// BinaryLinear_51719996178704
// MI455X (gfx1250) — matrix operand order repaired and checked by reading, not compiled, not run on hardware
//
#include <hip/hip_runtime.h>

typedef __bf16 bf16_t;
typedef __attribute__((ext_vector_type(16))) __bf16 v16bf;
typedef __attribute__((ext_vector_type(8)))  __bf16 v8bf;
typedef __attribute__((ext_vector_type(8)))  float  v8f;
typedef __attribute__((ext_vector_type(4)))  float  f32x4;

union V16 { v16bf v; v8bf h[2]; };

constexpr int BM = 128;     // workgroup M tile
constexpr int BN = 128;     // workgroup N tile
constexpr int BK = 32;      // K step (one WMMA depth)
constexpr int KDIM = 1024;
constexpr int NDIM = 1024;
constexpr int LDSS = 40;    // padded LDS row stride in bf16 elements (32 + 8)

__launch_bounds__(256)
__global__ void bnn_gemm_kernel(const float* __restrict__ x,
                                const float* __restrict__ w,
                                float* __restrict__ out)
{
    // double-buffered: hi/lo bf16 split of x tile + bf16 sign(W) tile (~60 KB)
    __shared__ __align__(16) bf16_t sAhi[2][BM * LDSS];
    __shared__ __align__(16) bf16_t sAlo[2][BM * LDSS];
    __shared__ __align__(16) bf16_t sB  [2][BN * LDSS];

    const int tid   = threadIdx.x;
    const int lane  = tid & 31;
    const int wv    = tid >> 5;          // 0..7
    const int waveM = wv >> 1;           // 0..3 : which 32-row slice
    const int waveN = wv & 1;            // 0..1 : which 64-col slice

    const int tileM = blockIdx.y * BM;
    const int tileN = blockIdx.x * BN;

    // cooperative loader coordinates: 8 threads x 4 floats cover one 32-k row
    const int lr = tid >> 3;             // 0..31
    const int lc = (tid & 7) * 4;        // 0,4,...,28

    // fragment coordinates (ISA 16-bit A/B layouts, wave32)
    const int khalf = lane >> 4;         // selects k sub-range
    const int lrow  = lane & 15;         // M for A-frag, N for B-frag / C

    v8f acc[2][4] = {};

    const float* Xb = x + (size_t)tileM * KDIM;
    const float* Wb = w + (size_t)tileN * KDIM;

    f32x4 xr[4], wr[4];                  // register staging (prefetch) buffers

    // convert current register staging into LDS buffer `b`
    auto stage = [&](int b) {
        #pragma unroll
        for (int rr = 0; rr < 4; ++rr) {
            const int r = lr + rr * 32;
            #pragma unroll
            for (int i = 0; i < 4; ++i) {
                float  f = xr[rr][i];
                bf16_t h = (bf16_t)f;                       // RNE
                sAhi[b][r * LDSS + lc + i] = h;
                sAlo[b][r * LDSS + lc + i] = (bf16_t)(f - (float)h);
                float  g = wr[rr][i];
                sB  [b][r * LDSS + lc + i] =
                    (bf16_t)((g > 0.0f) ? 1.0f : ((g < 0.0f) ? -1.0f : 0.0f));
            }
        }
    };

    // ---- prologue: load + stage tile k0 = 0 ----
    #pragma unroll
    for (int rr = 0; rr < 4; ++rr) {
        const int r = lr + rr * 32;
        xr[rr] = *(const f32x4*)(Xb + (size_t)r * KDIM + lc);
        wr[rr] = *(const f32x4*)(Wb + (size_t)r * KDIM + lc);
    }
    stage(0);
    __syncthreads();

    int buf = 0;
    for (int k0 = 0; k0 < KDIM; k0 += BK) {
        const bool has_next = (k0 + BK) < KDIM;

        // issue next tile's global loads FIRST: latency hides behind 16 WMMAs
        if (has_next) {
            #pragma unroll
            for (int rr = 0; rr < 4; ++rr) {
                const int r = lr + rr * 32;
                xr[rr] = *(const f32x4*)(Xb + (size_t)r * KDIM + k0 + BK + lc);
                wr[rr] = *(const f32x4*)(Wb + (size_t)r * KDIM + k0 + BK + lc);
            }
        }

        const bf16_t* pB  = sB[buf];
        const bf16_t* pHi = sAhi[buf];
        const bf16_t* pLo = sAlo[buf];

        // ---- B fragments: lane holds N = ns*16+lrow, K = khalf*8 + {0..7,16..23} ----
        V16 bfrag[4];
        #pragma unroll
        for (int ns = 0; ns < 4; ++ns) {
            const int n = waveN * 64 + ns * 16 + lrow;
            const v8bf* p = (const v8bf*)(pB + n * LDSS + khalf * 8);
            bfrag[ns].h[0] = p[0];
            bfrag[ns].h[1] = p[2];
        }

        // ---- A-hi fragments: lane holds M=lrow, K = khalf*8 + {0..7,16..23} ----
        V16 afrag[2];
        #pragma unroll
        for (int ms = 0; ms < 2; ++ms) {
            const int m = waveM * 32 + ms * 16 + lrow;
            const bf16_t* base = pHi + m * LDSS + khalf * 8;
            afrag[ms].h[0] = *(const v8bf*)(base);
            afrag[ms].h[1] = *(const v8bf*)(base + 16);
        }
        #pragma unroll
        for (int ms = 0; ms < 2; ++ms)
            #pragma unroll
            for (int ns = 0; ns < 4; ++ns)
                acc[ms][ns] = __builtin_amdgcn_wmma_f32_16x16x32_bf16(
                    false, afrag[ms].v, false, bfrag[ns].v,
                    (short)0, acc[ms][ns], false, false);

        // ---- A-lo fragments (residual pass, same accumulators) ----
        #pragma unroll
        for (int ms = 0; ms < 2; ++ms) {
            const int m = waveM * 32 + ms * 16 + lrow;
            const bf16_t* base = pLo + m * LDSS + khalf * 8;
            afrag[ms].h[0] = *(const v8bf*)(base);
            afrag[ms].h[1] = *(const v8bf*)(base + 16);
        }
        #pragma unroll
        for (int ms = 0; ms < 2; ++ms)
            #pragma unroll
            for (int ns = 0; ns < 4; ++ns)
                acc[ms][ns] = __builtin_amdgcn_wmma_f32_16x16x32_bf16(
                    false, afrag[ms].v, false, bfrag[ns].v,
                    (short)0, acc[ms][ns], false, false);

        // ---- stage next tile into the other buffer; single barrier/iter ----
        if (has_next) {
            stage(buf ^ 1);
            __syncthreads();
            buf ^= 1;
        }
    }

    // ---- store: C/D layout => lane: N = lrow, M = vgpr + 8*khalf ----
    #pragma unroll
    for (int ms = 0; ms < 2; ++ms) {
        const int mbase = tileM + waveM * 32 + ms * 16 + khalf * 8;
        #pragma unroll
        for (int ns = 0; ns < 4; ++ns) {
            const int n = tileN + waveN * 64 + ns * 16 + lrow;
            #pragma unroll
            for (int e = 0; e < 8; ++e)
                out[(size_t)(mbase + e) * NDIM + n] = acc[ms][ns][e];
        }
    }
}

extern "C" void kernel_launch(void* const* d_in, const int* in_sizes, int n_in,
                              void* d_out, int out_size, void* d_ws, size_t ws_size,
                              hipStream_t stream) {
    const float* x = (const float*)d_in[0];   // (32,4096,1024) fp32
    const float* w = (const float*)d_in[1];   // (1024,1024) fp32
    float* out = (float*)d_out;               // (32,4096,1024) fp32

    const long long M = (long long)in_sizes[0] / KDIM;   // 131072
    dim3 grid(NDIM / BN, (unsigned)(M / BM));            // (8, 1024): N fast so
                                                         // same-M blocks share x via L2
    bnn_gemm_kernel<<<grid, dim3(256), 0, stream>>>(x, w, out);
}
